// Encoder_Decoder_84928683311595
// MI455X (gfx1250) — matrix operand order repaired and checked by reading, not compiled, not run on hardware
//
#include <hip/hip_runtime.h>
#include <math.h>

// ---------------- types for WMMA ----------------
typedef __bf16 bf16;
typedef __attribute__((ext_vector_type(16))) __bf16 v16bf;
typedef __attribute__((ext_vector_type(8)))  __bf16 v8bf;
typedef __attribute__((ext_vector_type(8)))  float  v8f;

#define HDIM   256
#define BDIM   64
#define TSRC   64
#define TTGT   32
#define VOUT   32000
#define NTILE  (VOUT / 16)   // 2000 column tiles in the logits matrix

// =====================================================================
// RNN/attention GEMM:  C[64,N] = A1[64,K1]*B1^T + A2[64,K2]*B2^T + bias
// A row-major bf16 (lda), B = weight rows [N, K] bf16 (ldb) == ISA B-layout,
// f32 accumulate via v_wmma_f32_16x16x32_bf16.  wave -> M tile, block -> N tile.
// act: 0 = none, 1 = tanh.  C (f32) and Cbf (bf16) outputs optional.
// =====================================================================
__global__ __launch_bounds__(128) void gemm_bf16_wmma(
    const bf16* __restrict__ A1, const bf16* __restrict__ B1, int K1, int lda1, int ldb1,
    const bf16* __restrict__ A2, const bf16* __restrict__ B2, int K2, int lda2, int ldb2,
    const float* __restrict__ bias,
    float* __restrict__ C, bf16* __restrict__ Cbf, int N, int ldc, int act)
{
  const int tid  = threadIdx.x;
  const int wave = tid >> 5;
  const int lane = tid & 31;
  const int half = lane >> 4;     // 16-lane group
  const int ncol = lane & 15;     // B/C column in tile; also A row in tile
  const int mbase = wave * 16;    // 4 waves cover M = 64
  const int nbase = blockIdx.x * 16;
  const int n = nbase + ncol;

  v8f acc;
  const float bv = bias ? bias[n] : 0.0f;
#pragma unroll
  for (int r = 0; r < 8; ++r) acc[r] = bv;

  union V16 { v16bf v; v8bf h[2]; };

  {
    const bf16* arow = A1 + (size_t)(mbase + ncol) * lda1;
    const bf16* brow = B1 + (size_t)n * ldb1;
    for (int k0 = 0; k0 < K1; k0 += 32) {
      V16 a, b;
      a.h[0] = *(const v8bf*)(arow + k0 + half * 8);
      a.h[1] = *(const v8bf*)(arow + k0 + 16 + half * 8);
      b.h[0] = *(const v8bf*)(brow + k0 + half * 8);
      b.h[1] = *(const v8bf*)(brow + k0 + 16 + half * 8);
      acc = __builtin_amdgcn_wmma_f32_16x16x32_bf16(
          false, a.v, false, b.v, (short)0, acc, false, false);
    }
  }
  if (A2) {
    const bf16* arow = A2 + (size_t)(mbase + ncol) * lda2;
    const bf16* brow = B2 + (size_t)n * ldb2;
    for (int k0 = 0; k0 < K2; k0 += 32) {
      V16 a, b;
      a.h[0] = *(const v8bf*)(arow + k0 + half * 8);
      a.h[1] = *(const v8bf*)(arow + k0 + 16 + half * 8);
      b.h[0] = *(const v8bf*)(brow + k0 + half * 8);
      b.h[1] = *(const v8bf*)(brow + k0 + 16 + half * 8);
      acc = __builtin_amdgcn_wmma_f32_16x16x32_bf16(
          false, a.v, false, b.v, (short)0, acc, false, false);
    }
  }

  if (act == 1) {
#pragma unroll
    for (int r = 0; r < 8; ++r) acc[r] = tanhf(acc[r]);
  }

#pragma unroll
  for (int r = 0; r < 8; ++r) {
    const int m = mbase + half * 8 + r;
    if (C)   C[(size_t)m * ldc + n]   = acc[r];
    if (Cbf) Cbf[(size_t)m * ldc + n] = (bf16)acc[r];
  }
}

// =====================================================================
// Fused logits GEMM + per-tile log-softmax stats + target-logit capture.
//   logits[64, 32000] = A[64,256] * Wout[32000,256]^T + b_out
// wave -> N tile (each W_out element loaded exactly once), 4 M-tile
// accumulators per wave cover all 64 rows.  Epilogue: per row, 16-lane
// shfl_xor reduction -> (rowmax, sum exp) partial per tile; lane whose
// column equals the row's target token stores the target logit.
// Nothing 32000-wide is ever written to memory.
// =====================================================================
__global__ __launch_bounds__(128) void logits_fused_wmma(
    const bf16* __restrict__ A,        // [64, 256] bf16 (ht_new)
    const bf16* __restrict__ W,        // [32000, 256] bf16
    const float* __restrict__ bias,    // [32000]
    const int* __restrict__ tok_next,  // [64]
    float* __restrict__ part_max,      // [64 * NTILE]
    float* __restrict__ part_sum,      // [64 * NTILE]
    float* __restrict__ tgt_logit)     // [64]
{
  const int tid  = threadIdx.x;
  const int wave = tid >> 5;
  const int lane = tid & 31;
  const int half = lane >> 4;
  const int ncol = lane & 15;
  const int tile = blockIdx.x * 4 + wave;   // global N tile, 0..1999
  const int nbase = tile * 16;
  const int n = nbase + ncol;

  v8f acc[4];
  const float bv = bias[n];
#pragma unroll
  for (int mt = 0; mt < 4; ++mt)
#pragma unroll
    for (int r = 0; r < 8; ++r) acc[mt][r] = bv;

  union V16 { v16bf v; v8bf h[2]; };

  const bf16* wrow = W + (size_t)n * HDIM;
  for (int k0 = 0; k0 < HDIM; k0 += 32) {
    V16 b;
    b.h[0] = *(const v8bf*)(wrow + k0 + half * 8);   // B loaded ONCE per element
    b.h[1] = *(const v8bf*)(wrow + k0 + 16 + half * 8);
#pragma unroll
    for (int mt = 0; mt < 4; ++mt) {
      V16 a;
      const bf16* arow = A + (size_t)(mt * 16 + ncol) * HDIM;
      a.h[0] = *(const v8bf*)(arow + k0 + half * 8);
      a.h[1] = *(const v8bf*)(arow + k0 + 16 + half * 8);
      acc[mt] = __builtin_amdgcn_wmma_f32_16x16x32_bf16(
          false, a.v, false, b.v, (short)0, acc[mt], false, false);
    }
  }

  // Epilogue: per-row (16-lane) max / sum-exp partials + target capture.
#pragma unroll
  for (int mt = 0; mt < 4; ++mt) {
#pragma unroll
    for (int r = 0; r < 8; ++r) {
      const float v = acc[mt][r];
      const int row = mt * 16 + half * 8 + r;       // batch index b
      float mx = v;
      mx = fmaxf(mx, __shfl_xor(mx, 1, 16));
      mx = fmaxf(mx, __shfl_xor(mx, 2, 16));
      mx = fmaxf(mx, __shfl_xor(mx, 4, 16));
      mx = fmaxf(mx, __shfl_xor(mx, 8, 16));
      float e = expf(v - mx);
      e += __shfl_xor(e, 1, 16);
      e += __shfl_xor(e, 2, 16);
      e += __shfl_xor(e, 4, 16);
      e += __shfl_xor(e, 8, 16);
      if (ncol == 0) {
        part_max[(size_t)row * NTILE + tile] = mx;
        part_sum[(size_t)row * NTILE + tile] = e;
      }
      if (tok_next[row] == n) tgt_logit[row] = v;   // exactly one lane fires
    }
  }
}

// Combine NTILE partials per row -> masked NLL, accumulate per-step loss.
__global__ void logsumexp_combine(const float* __restrict__ part_max,
                                  const float* __restrict__ part_sum,
                                  const float* __restrict__ tgt_logit,
                                  const int* __restrict__ tok_next,
                                  float* __restrict__ step_num,
                                  float* __restrict__ step_den, int t) {
  int b = blockIdx.x, i = threadIdx.x;   // 64 x 256
  __shared__ float red[256];
  const float* pm = part_max + (size_t)b * NTILE;
  const float* ps = part_sum + (size_t)b * NTILE;
  float m = -1e30f;
  for (int k = i; k < NTILE; k += 256) m = fmaxf(m, pm[k]);
  red[i] = m; __syncthreads();
  for (int off = 128; off > 0; off >>= 1) { if (i < off) red[i] = fmaxf(red[i], red[i + off]); __syncthreads(); }
  float mx = red[0]; __syncthreads();
  float s = 0.0f;
  for (int k = i; k < NTILE; k += 256) s += ps[k] * expf(pm[k] - mx);
  red[i] = s; __syncthreads();
  for (int off = 128; off > 0; off >>= 1) { if (i < off) red[i] += red[i + off]; __syncthreads(); }
  if (i == 0) {
    int tok = tok_next[b];
    float nll = -(tgt_logit[b] - mx - logf(red[0]));
    float msk = (tok != 0) ? 1.0f : 0.0f;
    atomicAdd(&step_num[t], nll * msk);
    atomicAdd(&step_den[t], msk);
  }
}

// ---------------- small helper kernels ----------------
__global__ void cvt_f32_bf16(const float* __restrict__ s, bf16* __restrict__ d, int n) {
  int i = blockIdx.x * blockDim.x + threadIdx.x;
  if (i < n) d[i] = (bf16)s[i];
}
__global__ void bias_sum(const float* a, const float* b, float* o, int n) {
  int i = blockIdx.x * blockDim.x + threadIdx.x;
  if (i < n) o[i] = a[i] + b[i];
}
__global__ void zero_u32(unsigned int* p, int n) {
  int i = blockIdx.x * blockDim.x + threadIdx.x;
  if (i < n) p[i] = 0u;
}
__global__ void gather_embed(const float* __restrict__ embed, const int* __restrict__ toks,
                             bf16* __restrict__ xbf) {
  int b = blockIdx.x, j = threadIdx.x;            // 64 x 256
  int tok = toks[b];
  xbf[b * HDIM + j] = (bf16)embed[(size_t)tok * HDIM + j];
}

__device__ __forceinline__ float sigmoidf(float x) { return 1.0f / (1.0f + expf(-x)); }

// Encoder LSTM pointwise: faithful to source bug (cell input state always 0)
// + masked hidden update at padding tokens.
__global__ void enc_pointwise(const float* __restrict__ gates, const int* __restrict__ toks,
                              float* __restrict__ h, bf16* __restrict__ hbf,
                              float* __restrict__ c, float* __restrict__ list_hs_t) {
  int b = blockIdx.x, j = threadIdx.x;            // 64 x 256
  size_t gb = (size_t)b * 4 * HDIM;
  float ig = sigmoidf(gates[gb + j]);
  float og = sigmoidf(gates[gb + 3 * HDIM + j]);
  float gg = tanhf(gates[gb + 2 * HDIM + j]);
  float cn = ig * gg;                              // f * 0 + i*g (source bug)
  float hn = og * tanhf(cn);
  size_t idx = (size_t)b * HDIM + j;
  c[idx] = cn;
  float hs2 = (toks[b] != 0) ? hn : h[idx];
  h[idx] = hs2;
  hbf[idx] = (bf16)hs2;
  list_hs_t[idx] = hs2;
}

// Decoder LSTM pointwise: standard cell; recurrence keeps raw ht.
__global__ void dec_pointwise(const float* __restrict__ gates,
                              float* __restrict__ h, bf16* __restrict__ hbf,
                              float* __restrict__ c) {
  int b = blockIdx.x, j = threadIdx.x;
  size_t gb = (size_t)b * 4 * HDIM;
  float ig = sigmoidf(gates[gb + j]);
  float fg = sigmoidf(gates[gb + HDIM + j]);
  float gg = tanhf(gates[gb + 2 * HDIM + j]);
  float og = sigmoidf(gates[gb + 3 * HDIM + j]);
  size_t idx = (size_t)b * HDIM + j;
  float cn = fg * c[idx] + ig * gg;
  float hn = og * tanhf(cn);
  c[idx] = cn;
  h[idx] = hn;
  hbf[idx] = (bf16)hn;
}

// scores[t,b] = dot(ht[b,:], list_hs[t,b,:])
__global__ void attn_scores(const float* __restrict__ ht, const float* __restrict__ list_hs,
                            float* __restrict__ scores) {
  int t = blockIdx.x, b = blockIdx.y, i = threadIdx.x;  // (64,64) x 64
  __shared__ float red[64];
  const float* hv = ht + (size_t)b * HDIM;
  const float* sv = list_hs + ((size_t)t * BDIM + b) * HDIM;
  float s = 0.0f;
  for (int k = i; k < HDIM; k += 64) s += hv[k] * sv[k];
  red[i] = s; __syncthreads();
  for (int off = 32; off > 0; off >>= 1) { if (i < off) red[i] += red[i + off]; __syncthreads(); }
  if (i == 0) scores[t * BDIM + b] = red[0];
}

// softmax over t (64 vals) then context d[b,:] -> bf16
__global__ void softmax_ctx(const float* __restrict__ scores, const float* __restrict__ list_hs,
                            bf16* __restrict__ dbf) {
  int b = blockIdx.x, j = threadIdx.x;   // 64 x 256
  __shared__ float a[TSRC];
  if (j == 0) {
    float m = -1e30f;
    for (int t = 0; t < TSRC; ++t) m = fmaxf(m, scores[t * BDIM + b]);
    float s = 0.0f;
    for (int t = 0; t < TSRC; ++t) { float e = expf(scores[t * BDIM + b] - m); a[t] = e; s += e; }
    float inv = 1.0f / s;
    for (int t = 0; t < TSRC; ++t) a[t] *= inv;
  }
  __syncthreads();
  float acc = 0.0f;
  for (int t = 0; t < TSRC; ++t) acc += a[t] * list_hs[((size_t)t * BDIM + b) * HDIM + j];
  dbf[(size_t)b * HDIM + j] = (bf16)acc;
}

__global__ void finalize(const float* __restrict__ step_num, const float* __restrict__ step_den,
                         float* __restrict__ out) {
  if (threadIdx.x == 0) {
    float s = 0.0f;
    for (int t = 0; t < TTGT - 1; ++t) s += step_num[t] / fmaxf(step_den[t], 1.0f);
    out[0] = s;
  }
}

// =====================================================================
extern "C" void kernel_launch(void* const* d_in, const int* in_sizes, int n_in,
                              void* d_out, int out_size, void* d_ws, size_t ws_size,
                              hipStream_t stream) {
  const int*   input_lines  = (const int*)  d_in[0];
  const int*   target_lines = (const int*)  d_in[1];
  const float* embed_input  = (const float*)d_in[2];
  const float* embed_target = (const float*)d_in[3];
  const float* Wih_e = (const float*)d_in[4];
  const float* Whh_e = (const float*)d_in[5];
  const float* bih_e = (const float*)d_in[6];
  const float* bhh_e = (const float*)d_in[7];
  const float* Wih_d = (const float*)d_in[8];
  const float* Whh_d = (const float*)d_in[9];
  const float* bih_d = (const float*)d_in[10];
  const float* bhh_d = (const float*)d_in[11];
  const float* W_attn = (const float*)d_in[12];
  const float* b_attn = (const float*)d_in[13];
  const float* W_out  = (const float*)d_in[14];
  const float* b_out  = (const float*)d_in[15];
  float* out = (float*)d_out;

  // ---- scratch layout (256B aligned slabs) ----
  char* ws = (char*)d_ws;
  size_t off = 0;
  auto take = [&](size_t bytes) -> void* {
    void* p = ws + off;
    off += (bytes + 255) & ~(size_t)255;
    return p;
  };
  bf16* Wih_e_bf = (bf16*)take((size_t)4 * HDIM * HDIM * 2);
  bf16* Whh_e_bf = (bf16*)take((size_t)4 * HDIM * HDIM * 2);
  bf16* Wih_d_bf = (bf16*)take((size_t)4 * HDIM * HDIM * 2);
  bf16* Whh_d_bf = (bf16*)take((size_t)4 * HDIM * HDIM * 2);
  bf16* Wattn_bf = (bf16*)take((size_t)HDIM * 2 * HDIM * 2);
  bf16* Wout_bf  = (bf16*)take((size_t)VOUT * HDIM * 2);
  float* bsum_e  = (float*)take(4 * HDIM * 4);
  float* bsum_d  = (float*)take(4 * HDIM * 4);
  bf16*  x_bf    = (bf16*) take((size_t)BDIM * HDIM * 2);
  bf16*  h_bf    = (bf16*) take((size_t)BDIM * HDIM * 2);
  float* h_f     = (float*)take((size_t)BDIM * HDIM * 4);
  float* c_f     = (float*)take((size_t)BDIM * HDIM * 4);
  float* gates   = (float*)take((size_t)BDIM * 4 * HDIM * 4);
  float* list_hs = (float*)take((size_t)TSRC * BDIM * HDIM * 4);
  float* scores  = (float*)take((size_t)TSRC * BDIM * 4);
  bf16*  d_bf    = (bf16*) take((size_t)BDIM * HDIM * 2);
  bf16*  htn_bf  = (bf16*) take((size_t)BDIM * HDIM * 2);
  float* part_max = (float*)take((size_t)BDIM * NTILE * 4);
  float* part_sum = (float*)take((size_t)BDIM * NTILE * 4);
  float* tgt_logit = (float*)take(BDIM * 4);
  float* step_num = (float*)take(64 * 4);
  float* step_den = step_num + 32;
  (void)ws_size; (void)in_sizes; (void)n_in; (void)out_size;

  // ---- one-time weight conversion to bf16 (layout preserved: [N, K]) ----
  const int WELEM = 4 * HDIM * HDIM;                       // 262144
  cvt_f32_bf16<<<(WELEM + 255) / 256, 256, 0, stream>>>(Wih_e, Wih_e_bf, WELEM);
  cvt_f32_bf16<<<(WELEM + 255) / 256, 256, 0, stream>>>(Whh_e, Whh_e_bf, WELEM);
  cvt_f32_bf16<<<(WELEM + 255) / 256, 256, 0, stream>>>(Wih_d, Wih_d_bf, WELEM);
  cvt_f32_bf16<<<(WELEM + 255) / 256, 256, 0, stream>>>(Whh_d, Whh_d_bf, WELEM);
  cvt_f32_bf16<<<(HDIM * 2 * HDIM + 255) / 256, 256, 0, stream>>>(W_attn, Wattn_bf, HDIM * 2 * HDIM);
  cvt_f32_bf16<<<((size_t)VOUT * HDIM + 255) / 256, 256, 0, stream>>>(W_out, Wout_bf, VOUT * HDIM);
  bias_sum<<<(4 * HDIM + 255) / 256, 256, 0, stream>>>(bih_e, bhh_e, bsum_e, 4 * HDIM);
  bias_sum<<<(4 * HDIM + 255) / 256, 256, 0, stream>>>(bih_d, bhh_d, bsum_d, 4 * HDIM);

  // ---- state init ----
  zero_u32<<<(BDIM * HDIM + 255) / 256, 256, 0, stream>>>((unsigned int*)h_f, BDIM * HDIM);
  zero_u32<<<(BDIM * HDIM + 255) / 256, 256, 0, stream>>>((unsigned int*)c_f, BDIM * HDIM);
  zero_u32<<<(BDIM * HDIM / 2 + 255) / 256, 256, 0, stream>>>((unsigned int*)h_bf, BDIM * HDIM / 2);
  zero_u32<<<1, 64, 0, stream>>>((unsigned int*)step_num, 64);

  // ---- encoder: 64 sequential steps ----
  for (int t = 0; t < TSRC; ++t) {
    gather_embed<<<BDIM, HDIM, 0, stream>>>(embed_input, input_lines + t * BDIM, x_bf);
    gemm_bf16_wmma<<<4 * HDIM / 16, 128, 0, stream>>>(
        x_bf, Wih_e_bf, HDIM, HDIM, HDIM,
        h_bf, Whh_e_bf, HDIM, HDIM, HDIM,
        bsum_e, gates, nullptr, 4 * HDIM, 4 * HDIM, 0);
    enc_pointwise<<<BDIM, HDIM, 0, stream>>>(gates, input_lines + t * BDIM,
                                             h_f, h_bf, c_f,
                                             list_hs + (size_t)t * BDIM * HDIM);
  }

  // ---- decoder: 31 sequential steps ----
  for (int t = 0; t < TTGT - 1; ++t) {
    gather_embed<<<BDIM, HDIM, 0, stream>>>(embed_target, target_lines + t * BDIM, x_bf);
    gemm_bf16_wmma<<<4 * HDIM / 16, 128, 0, stream>>>(
        x_bf, Wih_d_bf, HDIM, HDIM, HDIM,
        h_bf, Whh_d_bf, HDIM, HDIM, HDIM,
        bsum_d, gates, nullptr, 4 * HDIM, 4 * HDIM, 0);
    dec_pointwise<<<BDIM, HDIM, 0, stream>>>(gates, h_f, h_bf, c_f);

    attn_scores<<<dim3(TSRC, BDIM), 64, 0, stream>>>(h_f, list_hs, scores);
    softmax_ctx<<<BDIM, HDIM, 0, stream>>>(scores, list_hs, d_bf);

    // ht_new = tanh(cat[d, ht] @ W_attn^T + b_attn)  -> concat split into 2 K-passes
    gemm_bf16_wmma<<<HDIM / 16, 128, 0, stream>>>(
        d_bf, Wattn_bf,        HDIM, HDIM, 2 * HDIM,
        h_bf, Wattn_bf + HDIM, HDIM, HDIM, 2 * HDIM,
        b_attn, nullptr, htn_bf, HDIM, HDIM, 1);

    // fused logits GEMM + log-softmax partials (500 blocks, W_out read once)
    logits_fused_wmma<<<NTILE / 4, 128, 0, stream>>>(
        htn_bf, Wout_bf, b_out, target_lines + (t + 1) * BDIM,
        part_max, part_sum, tgt_logit);

    logsumexp_combine<<<BDIM, 256, 0, stream>>>(part_max, part_sum, tgt_logit,
                                                target_lines + (t + 1) * BDIM,
                                                step_num, step_den, t);
  }

  finalize<<<1, 32, 0, stream>>>(step_num, step_den, out);
}
